// AttentionPointnet_16655883174795
// MI455X (gfx1250) — compile-verified, matrix operand order checked by reading, not run on hardware
//
#include <hip/hip_runtime.h>

// ---------------- problem constants ----------------
#define BB    8
#define TT    4096
#define DD    3
#define HH    128
#define NBLK  6
#define KK    20
#define CDIMM 128
#define CTXC  7
#define NPTS  (BB*TT)

typedef __attribute__((ext_vector_type(16))) _Float16 v16h;
typedef __attribute__((ext_vector_type(8)))  _Float16 v8h;
typedef __attribute__((ext_vector_type(8)))  float    v8f;

__device__ __forceinline__ v8f zero8() {
    v8f z;
#pragma unroll
    for (int i = 0; i < 8; ++i) z[i] = 0.f;
    return z;
}

// ------------------------------------------------------------------
// WMMA fragment loaders (16x16x32 f16, wave32).
// 16-bit operand layout (ISA 7.12.2): lane L (0..15 / 16..31), halves
// 0..7 -> K = klo+0..7, halves 8..15 -> K = klo+16..23, klo = 8*(L>=16).
// A rows: row-major f16 (LDS or global), stride rs halves. B columns:
// pre-transposed f16 weights Wt[n][k] (k contiguous) in global/L2.
// ------------------------------------------------------------------
template<bool RELU>
__device__ __forceinline__ v16h load_afrag(const _Float16* base, int rs, int k0) {
    const int lane = threadIdx.x & 31;
    const int m    = lane & 15;
    const int klo  = (lane & 16) ? 8 : 0;
    const _Float16* ptr = base + m * rs + k0 + klo;
    v8h x0 = *(const v8h*)(ptr);
    v8h x1 = *(const v8h*)(ptr + 16);
    if (RELU) {
#pragma unroll
        for (int e = 0; e < 8; ++e) {
            x0[e] = (x0[e] > (_Float16)0.f) ? x0[e] : (_Float16)0.f;
            x1[e] = (x1[e] > (_Float16)0.f) ? x1[e] : (_Float16)0.f;
        }
    }
    return __builtin_shufflevector(x0, x1, 0,1,2,3,4,5,6,7,8,9,10,11,12,13,14,15);
}

__device__ __forceinline__ v16h load_bfrag(const _Float16* wt, int Kd, int n0, int k0) {
    const int lane = threadIdx.x & 31;
    const int n    = n0 + (lane & 15);
    const int klo  = (lane & 16) ? 8 : 0;
    const _Float16* ptr = wt + (size_t)n * Kd + k0 + klo;
    v8h x0 = *(const v8h*)(ptr);
    v8h x1 = *(const v8h*)(ptr + 16);
    return __builtin_shufflevector(x0, x1, 0,1,2,3,4,5,6,7,8,9,10,11,12,13,14,15);
}

// C += A(16xKD row-major f16) * B(KDx128 from f16 W^T), f32 accum.
template<int KD, bool RELU>
__device__ __forceinline__ void gemm_16xH(v8f acc[8], const _Float16* aBase, int aStride,
                                          const _Float16* wt) {
#pragma unroll
    for (int kk = 0; kk < KD; kk += 32) {
        v16h a = load_afrag<RELU>(aBase, aStride, kk);
#pragma unroll
        for (int n = 0; n < 8; ++n) {
            v16h b = load_bfrag(wt, KD, n * 16, kk);
            acc[n] = __builtin_amdgcn_wmma_f32_16x16x32_f16(
                false, a, false, b, (short)0, acc[n], false, false);
        }
    }
}

// ------------------------------------------------------------------
// Weight convert + transpose: src [nmat][Kd][H] f32 -> dst [nmat][H][Kd] f16
// ------------------------------------------------------------------
__global__ void w_to_f16t_kernel(const float* __restrict__ src, _Float16* __restrict__ dst,
                                 int Kd, int total) {
    int i = blockIdx.x * 256 + threadIdx.x;
    if (i >= total) return;
    int mk  = Kd * HH;
    int mat = i / mk;
    int rem = i - mat * mk;
    int n   = rem / Kd;
    int k   = rem - n * Kd;
    dst[i] = (_Float16)src[(size_t)mat * mk + (size_t)k * HH + n];
}

// ------------------------------------------------------------------
// net0 = p @ W_pos + b_pos   (f32 + f16 mirror)
// ------------------------------------------------------------------
__global__ void init_net_kernel(const float* __restrict__ p, const float* __restrict__ Wpos,
                                const float* __restrict__ bpos, float* __restrict__ net,
                                _Float16* __restrict__ net16) {
    int i = blockIdx.x * 256 + threadIdx.x;
    int t = i >> 7, c = i & 127;
    float acc = bpos[c];
#pragma unroll
    for (int d = 0; d < DD; ++d) acc += p[t * 3 + d] * Wpos[d * HH + c];
    net[i]   = acc;
    net16[i] = (_Float16)acc;
}

// ------------------------------------------------------------------
// KNN + context. One thread per point, LDS tiles of 256 candidates,
// register insertion-sort top-20 (fully unrolled -> stays in VGPRs).
// idx stores GLOBAL row (b*T + j). ctx: [g][k][7] = {dist, p_j, p_i}.
// ------------------------------------------------------------------
__global__ __launch_bounds__(256)
void knn_ctx_kernel(const float* __restrict__ p, int* __restrict__ idx_out,
                    float* __restrict__ ctx_out) {
    __shared__ float sP[256][3];
    const int tid  = threadIdx.x;
    const int g    = blockIdx.x * 256 + tid;
    const int base = (g / TT) * TT;
    const float xi = p[g * 3 + 0], yi = p[g * 3 + 1], zi = p[g * 3 + 2];

    float bd[KK];
    int   bi[KK];
#pragma unroll
    for (int q = 0; q < KK; ++q) { bd[q] = 3.0e38f; bi[q] = base; }

    for (int jt = 0; jt < TT; jt += 256) {
        __syncthreads();
        const int jl = base + jt + tid;
        sP[tid][0] = p[jl * 3 + 0];
        sP[tid][1] = p[jl * 3 + 1];
        sP[tid][2] = p[jl * 3 + 2];
        __syncthreads();
        for (int jj = 0; jj < 256; ++jj) {
            float dx = xi - sP[jj][0], dy = yi - sP[jj][1], dz = zi - sP[jj][2];
            float d2 = dx * dx + dy * dy + dz * dz;
            if (d2 < bd[KK - 1]) {
                bd[KK - 1] = d2; bi[KK - 1] = base + jt + jj;
#pragma unroll
                for (int q = KK - 1; q > 0; --q) {
                    if (bd[q] < bd[q - 1]) {
                        float td = bd[q]; bd[q] = bd[q - 1]; bd[q - 1] = td;
                        int   ti = bi[q]; bi[q] = bi[q - 1]; bi[q - 1] = ti;
                    }
                }
            }
        }
    }
#pragma unroll
    for (int q = 0; q < KK; ++q) {
        int jg = bi[q];
        idx_out[(size_t)g * KK + q] = jg;
        float dis = sqrtf(fmaxf(bd[q], 1e-12f));
        float* c = ctx_out + ((size_t)g * KK + q) * CTXC;
        c[0] = dis;
        c[1] = p[jg * 3 + 0]; c[2] = p[jg * 3 + 1]; c[3] = p[jg * 3 + 2];
        c[4] = xi; c[5] = yi; c[6] = zi;
    }
}

// ------------------------------------------------------------------
// One residual attention block.
// WG = 128 threads (4 waves), 64 points. Phase A: softmax attention
// pooling, gathering the f16 net mirror (halves HBM gather traffic).
// Phase B: 4 chained WMMA GEMMs (Wo, W0, W1, Ws).
// net_out = x@Ws + fc1(relu(fc0(relu(x)))) + (add_skip ? net_in : 0)
// ------------------------------------------------------------------
__global__ __launch_bounds__(128)
void block_kernel(const float* __restrict__ net_in, const _Float16* __restrict__ net16_in,
                  float* __restrict__ net_out, _Float16* __restrict__ net16_out,
                  const int* __restrict__ idx, const float* __restrict__ ctx,
                  const _Float16* __restrict__ W0t, const float* __restrict__ b0,
                  const _Float16* __restrict__ W1t, const float* __restrict__ b1,
                  const _Float16* __restrict__ Wst,
                  const _Float16* __restrict__ Wot, const float* __restrict__ bo,
                  const float* __restrict__ Wc,  const float* __restrict__ bc,
                  int add_skip) {
    __shared__ _Float16 sX[64][2 * HH];  // x = [net | att] per point, f16
    __shared__ _Float16 sH[64][HH];      // att-input, then hidden h

    const int tid  = threadIdx.x;
    const int row0 = blockIdx.x * 64;

    // ---------- Phase A: attention pooling (4 threads x 32 ch per point) ----------
    const float wc0 = Wc[0], wc1 = Wc[1], wc2 = Wc[2], wc3 = Wc[3];
    const float wc4 = Wc[4], wc5 = Wc[5], wc6 = Wc[6], bcv = bc[0];
#pragma unroll
    for (int pp = 0; pp < 2; ++pp) {
        const int lp  = pp * 32 + (tid >> 2);
        const int g   = row0 + lp;
        const int ch0 = (tid & 3) * 32;

        float w[KK];
        float mx = -3.0e38f;
#pragma unroll
        for (int k = 0; k < KK; ++k) {
            const float* c = ctx + ((size_t)g * KK + k) * CTXC;
            float s = bcv + c[0] * wc0 + c[1] * wc1 + c[2] * wc2 + c[3] * wc3 +
                      c[4] * wc4 + c[5] * wc5 + c[6] * wc6;
            w[k] = s;
            mx = fmaxf(mx, s);
        }
        float sum = 0.f;
#pragma unroll
        for (int k = 0; k < KK; ++k) { w[k] = __expf(w[k] - mx); sum += w[k]; }
        const float inv = 1.f / sum;

        float acc[32];
#pragma unroll
        for (int c = 0; c < 32; ++c) acc[c] = 0.f;
#pragma unroll
        for (int k = 0; k < KK; ++k) {
            const float wk = w[k] * inv;
            const v8h* nb = (const v8h*)(net16_in + (size_t)idx[(size_t)g * KK + k] * HH + ch0);
#pragma unroll
            for (int c8 = 0; c8 < 4; ++c8) {
                v8h v = nb[c8];
#pragma unroll
                for (int e = 0; e < 8; ++e) acc[c8 * 8 + e] += wk * (float)v[e];
            }
        }
        // stage att input (pre-Wo) and x's left half (straight f16 copy)
#pragma unroll
        for (int c = 0; c < 32; ++c) sH[lp][ch0 + c] = (_Float16)acc[c];
        const v8h* own = (const v8h*)(net16_in + (size_t)g * HH + ch0);
#pragma unroll
        for (int c8 = 0; c8 < 4; ++c8) *(v8h*)&sX[lp][ch0 + c8 * 8] = own[c8];
    }
    __syncthreads();

    // ---------- Phase B: WMMA GEMMs, 16-row tile per wave ----------
    const int wv   = tid >> 5;
    const int lane = tid & 31;
    const int mrow = wv * 16;
    const int col  = lane & 15;
    const int rsel = (lane & 16) ? 8 : 0;

    v8f acc[8];

    // GEMM1: att = attIn @ Wo^T + bo  -> right half of x
#pragma unroll
    for (int n = 0; n < 8; ++n) acc[n] = zero8();
    gemm_16xH<HH, false>(acc, &sH[mrow][0], HH, Wot);
#pragma unroll
    for (int n = 0; n < 8; ++n) {
        const int cc = n * 16 + col;
        const float bias = bo[cc];
#pragma unroll
        for (int r = 0; r < 8; ++r)
            sX[mrow + r + rsel][HH + cc] = (_Float16)(acc[n][r] + bias);
    }
    __syncthreads();

    // GEMM2: h = relu( relu(x) @ W0^T + b0 )
#pragma unroll
    for (int n = 0; n < 8; ++n) acc[n] = zero8();
    gemm_16xH<2 * HH, true>(acc, &sX[mrow][0], 2 * HH, W0t);
#pragma unroll
    for (int n = 0; n < 8; ++n) {
        const int cc = n * 16 + col;
        const float bias = b0[cc];
#pragma unroll
        for (int r = 0; r < 8; ++r)
            sH[mrow + r + rsel][cc] = (_Float16)fmaxf(acc[n][r] + bias, 0.f);
    }
    __syncthreads();

    // GEMM3 + GEMM4: out = h @ W1^T + x @ Ws^T   (+ b1 + skip in epilogue)
#pragma unroll
    for (int n = 0; n < 8; ++n) acc[n] = zero8();
    gemm_16xH<HH, false>(acc, &sH[mrow][0], HH, W1t);
    gemm_16xH<2 * HH, false>(acc, &sX[mrow][0], 2 * HH, Wst);
#pragma unroll
    for (int n = 0; n < 8; ++n) {
        const int cc = n * 16 + col;
        const float bias = b1[cc];
#pragma unroll
        for (int r = 0; r < 8; ++r) {
            const int rr = row0 + mrow + r + rsel;
            float o = acc[n][r] + bias;
            if (add_skip) o += net_in[(size_t)rr * HH + cc];
            net_out[(size_t)rr * HH + cc]   = o;
            net16_out[(size_t)rr * HH + cc] = (_Float16)o;
        }
    }
}

// ------------------------------------------------------------------
// out = net @ W_c^T + b_c  (128x128, WMMA). A-fragments built directly
// from the global f16 net mirror (already row-major f16) — no staging.
// ------------------------------------------------------------------
__global__ __launch_bounds__(128)
void out_proj_kernel(const _Float16* __restrict__ net16, const _Float16* __restrict__ Wct,
                     const float* __restrict__ bcst, float* __restrict__ out) {
    const int tid  = threadIdx.x;
    const int row0 = blockIdx.x * 64;
    const int wv = tid >> 5, lane = tid & 31, mrow = wv * 16;
    const int col = lane & 15, rsel = (lane & 16) ? 8 : 0;
    v8f acc[8];
#pragma unroll
    for (int n = 0; n < 8; ++n) acc[n] = zero8();
    gemm_16xH<HH, false>(acc, net16 + (size_t)(row0 + mrow) * HH, HH, Wct);
#pragma unroll
    for (int n = 0; n < 8; ++n) {
        const int cc = n * 16 + col;
        const float bias = bcst[cc];
#pragma unroll
        for (int r = 0; r < 8; ++r)
            out[(size_t)(row0 + mrow + r + rsel) * CDIMM + cc] = acc[n][r] + bias;
    }
}

// ------------------------------------------------------------------
extern "C" void kernel_launch(void* const* d_in, const int* in_sizes, int n_in,
                              void* d_out, int out_size, void* d_ws, size_t ws_size,
                              hipStream_t stream) {
    (void)in_sizes; (void)n_in; (void)out_size; (void)ws_size;
    const float* p    = (const float*)d_in[0];
    const float* Wpos = (const float*)d_in[1];
    const float* bpos = (const float*)d_in[2];
    const float* W0   = (const float*)d_in[3];
    const float* b0   = (const float*)d_in[4];
    const float* W1   = (const float*)d_in[5];
    const float* b1   = (const float*)d_in[6];
    const float* Wsm  = (const float*)d_in[7];
    const float* Wc   = (const float*)d_in[8];
    const float* bc   = (const float*)d_in[9];
    const float* Wo   = (const float*)d_in[10];
    const float* bo   = (const float*)d_in[11];
    const float* Wcf  = (const float*)d_in[12];
    const float* bcf  = (const float*)d_in[13];
    float* out = (float*)d_out;

    char* ws = (char*)d_ws;
    size_t cur = 0;
    auto alloc = [&](size_t bytes) -> char* {
        char* r = ws + cur;
        cur = (cur + bytes + 255) & ~(size_t)255;
        return r;
    };

    int*      idxb   = (int*)alloc((size_t)NPTS * KK * sizeof(int));
    float*    ctxb   = (float*)alloc((size_t)NPTS * KK * CTXC * sizeof(float));
    float*    netA   = (float*)alloc((size_t)NPTS * HH * sizeof(float));
    float*    netB   = (float*)alloc((size_t)NPTS * HH * sizeof(float));
    _Float16* netA16 = (_Float16*)alloc((size_t)NPTS * HH * sizeof(_Float16));
    _Float16* netB16 = (_Float16*)alloc((size_t)NPTS * HH * sizeof(_Float16));
    _Float16* W0t    = (_Float16*)alloc((size_t)NBLK * 2 * HH * HH * sizeof(_Float16));
    _Float16* W1t    = (_Float16*)alloc((size_t)NBLK * HH * HH * sizeof(_Float16));
    _Float16* Wst    = (_Float16*)alloc((size_t)NBLK * 2 * HH * HH * sizeof(_Float16));
    _Float16* Wot    = (_Float16*)alloc((size_t)NBLK * HH * HH * sizeof(_Float16));
    _Float16* Wct    = (_Float16*)alloc((size_t)HH * CDIMM * sizeof(_Float16));

    // 1) weight convert + transpose to f16 (k-contiguous per output col)
    {
        int t0 = NBLK * 2 * HH * HH;   // W0, Ws  (Kd = 256)
        int t1 = NBLK * HH * HH;       // W1, Wo  (Kd = 128)
        int t2 = HH * CDIMM;           // W_c     (Kd = 128)
        w_to_f16t_kernel<<<(t0 + 255) / 256, 256, 0, stream>>>(W0,  W0t, 2 * HH, t0);
        w_to_f16t_kernel<<<(t1 + 255) / 256, 256, 0, stream>>>(W1,  W1t, HH,     t1);
        w_to_f16t_kernel<<<(t0 + 255) / 256, 256, 0, stream>>>(Wsm, Wst, 2 * HH, t0);
        w_to_f16t_kernel<<<(t1 + 255) / 256, 256, 0, stream>>>(Wo,  Wot, HH,     t1);
        w_to_f16t_kernel<<<(t2 + 255) / 256, 256, 0, stream>>>(Wcf, Wct, HH,     t2);
    }

    // 2) net0 = p @ W_pos + b_pos  (f32 + f16 mirror)
    init_net_kernel<<<(NPTS * HH) / 256, 256, 0, stream>>>(p, Wpos, bpos, netA, netA16);

    // 3) KNN + context
    knn_ctx_kernel<<<NPTS / 256, 256, 0, stream>>>(p, idxb, ctxb);

    // 4) 6 residual attention blocks (ping-pong; last_net == prev net)
    for (int i = 0; i < NBLK; ++i) {
        const float*    nin   = (i & 1) ? netB   : netA;
        const _Float16* nin16 = (i & 1) ? netB16 : netA16;
        float*          nout  = (i & 1) ? netA   : netB;
        _Float16*       nout16= (i & 1) ? netA16 : netB16;
        block_kernel<<<NPTS / 64, 128, 0, stream>>>(
            nin, nin16, nout, nout16, idxb, ctxb,
            W0t + (size_t)i * 2 * HH * HH, b0 + i * HH,
            W1t + (size_t)i * HH * HH,     b1 + i * HH,
            Wst + (size_t)i * 2 * HH * HH,
            Wot + (size_t)i * HH * HH,     bo + i * HH,
            Wc + i * CTXC, bc + i,
            (i > 0) ? 1 : 0);
    }

    // 5) output projection (NBLK even -> final net is in netA16)
    const _Float16* nfin16 = (NBLK & 1) ? netB16 : netA16;
    out_proj_kernel<<<NPTS / 64, 128, 0, stream>>>(nfin16, Wct, bcf, out);
}
